// KoopmanOperator_28870770163733
// MI455X (gfx1250) — matrix operand order repaired and checked by reading, not compiled, not run on hardware
//
#include <hip/hip_runtime.h>

// ---------------------------------------------------------------------------
// Koopman operator recurrence on MI455X (gfx1250), wave32 + WMMA.
//   per step:  h = tanh(y@W1 + b1) @ W2 + b2 ;  y <- rot(exp/cos/sin(h)) * y
//   B=4096 rows, K=64, T=256.  Each wave owns a 16-row M-tile for all T steps.
//
// Layouts (per ISA 7.12.2, wave32):
//   A  f16 16x32 : element e <-> K = 16*(e>>3) + 8*g + (e&7), row = lane&15
//   B  f16 32x16 : element e <-> K = 16*(e>>3) + 8*g + (e&7), col = lane&15
//   C/D f32 16x16: reg r     <-> row = 8*g + r,               col = lane&15
// State y kept in A-operand order (f32) so complex pairs (q,q+1) are lane-local.
// Biases folded: b1 into tanh (exp(2x+2b)), b2 into the h staging store, so
// WMMA C starts from a single loop-invariant zero vector (no per-step splats).
// ---------------------------------------------------------------------------

typedef __attribute__((ext_vector_type(16))) _Float16 v16h;
typedef __attribute__((ext_vector_type(8)))  float    v8f;

#define B_SZ     4096
#define K_SZ     64
#define T_STEPS  256
#define DT       0.01f
#define WAVES    4           // waves per block (independent 16-row tiles)

__device__ __forceinline__ float fast_tanh_biased(float x, float two_b) {
  // tanh(x + b) = 1 - 2/(exp(2x + 2b) + 1)   -> v_fma + v_exp_f32 + v_rcp_f32
  float ex = __expf(__builtin_fmaf(2.0f, x, two_b));
  return 1.0f - 2.0f * __builtin_amdgcn_rcpf(ex + 1.0f);
}

__device__ __forceinline__ v8f wmma_f16(v16h a, v16h b, v8f c) {
  return __builtin_amdgcn_wmma_f32_16x16x32_f16(
      /*neg_a=*/false, a, /*neg_b=*/false, b,
      /*c_mod=*/(short)0, c, /*reuse_a=*/false, /*reuse_b=*/false);
}

__global__ __launch_bounds__(32 * WAVES)
void koopman_kernel(const float* __restrict__ x,
                    const float* __restrict__ W1,
                    const float* __restrict__ b1,
                    const float* __restrict__ W2,
                    const float* __restrict__ b2,
                    float* __restrict__ out) {
  // W2 B-operand tiles, pre-swizzled into reader order: tile(kt,nt) -> 512 halves,
  // lane L reads 16 contiguous halves at [tile*512 + L*16].
  __shared__ __align__(32) _Float16 w2lds[8 * 512];        // 8 KB
  // Per-wave D->A transpose staging (f16 for tanh(t), f32 for h; reused in-order).
  __shared__ __align__(16) float stage[WAVES * 1024];      // 16 KB

  const int tid  = threadIdx.x;
  const int w    = tid >> 5;
  const int lane = tid & 31;
  const int c    = lane & 15;        // D-layout column / A-layout row
  const int g    = lane >> 4;        // lane group
  const int myRow = (blockIdx.x * WAVES + w) * 16 + c;

  // ---- W1 B-operand tiles resident in VGPRs: element e <-> K = 32*kt+16*(e>>3)+8*g+(e&7) ----
  v16h B1[2][4];
#pragma unroll
  for (int kt = 0; kt < 2; ++kt)
#pragma unroll
    for (int nt = 0; nt < 4; ++nt) {
      v16h tv;
#pragma unroll
      for (int e = 0; e < 16; ++e)
        tv[e] = (_Float16)W1[(32 * kt + 16 * (e >> 3) + 8 * g + (e & 7)) * K_SZ + 16 * nt + c];
      B1[kt][nt] = tv;
    }

  // ---- W2 into LDS in B-operand reader order (wave 0 writes, all read) ----
  if (w == 0) {
#pragma unroll
    for (int kt = 0; kt < 2; ++kt)
#pragma unroll
      for (int nt = 0; nt < 4; ++nt) {
        v16h tv;
#pragma unroll
        for (int e = 0; e < 16; ++e)
          tv[e] = (_Float16)W2[(32 * kt + 16 * (e >> 3) + 8 * g + (e & 7)) * K_SZ + 16 * nt + c];
        *(v16h*)&w2lds[((kt * 4 + nt) * 32 + lane) * 16] = tv;
      }
  }

  // ---- biases (per D-layout n-tile, broadcast over rows); b1 pre-doubled ----
  float tb1[4], b2v[4];
#pragma unroll
  for (int t = 0; t < 4; ++t) {
    tb1[t] = 2.0f * b1[16 * t + c];
    b2v[t] = b2[16 * t + c];
  }

  // ---- state y in A-operand order: q=0..31, col(q) = 16*(q>>3) + 8*g + (q&7) ----
  float y[32];
  {
    const float* xp = x + (size_t)myRow * K_SZ;
#pragma unroll
    for (int t = 0; t < 4; ++t) {
      float4 v0 = *(const float4*)(xp + 16 * t + 8 * g);
      float4 v1 = *(const float4*)(xp + 16 * t + 8 * g + 4);
      y[8 * t + 0] = v0.x; y[8 * t + 1] = v0.y; y[8 * t + 2] = v0.z; y[8 * t + 3] = v0.w;
      y[8 * t + 4] = v1.x; y[8 * t + 5] = v1.y; y[8 * t + 6] = v1.z; y[8 * t + 7] = v1.w;
    }
  }
  __syncthreads();

  float*     sw  = stage + w * 1024;      // 4 KB per wave
  _Float16*  swh = (_Float16*)sw;

  // single loop-invariant zero C operand
  v8f zacc;
#pragma unroll
  for (int i = 0; i < 8; ++i) zacc[i] = 0.0f;

  const int dstBase = 8 * g + 16 * (c >> 3);   // + r -> staging dest "lane"
  const int sub     = c & 7;

  for (int step = 0; step < T_STEPS; ++step) {
    // ---- A operands for GEMM1: pure per-lane f32->f16 convert ----
    v16h a0, a1;
#pragma unroll
    for (int e = 0; e < 16; ++e) {
      a0[e] = (_Float16)y[e];
      a1[e] = (_Float16)y[16 + e];
    }

    // ---- GEMM1: t = y@W1  (8x v_wmma_f32_16x16x32_f16, C=0) ----
    v8f acc[4];
#pragma unroll
    for (int nt = 0; nt < 4; ++nt) {
      v8f a = wmma_f16(a0, B1[0][nt], zacc);
      a     = wmma_f16(a1, B1[1][nt], a);
      acc[nt] = a;
    }

    // ---- tanh(t + b1), then D->A transpose through LDS as f16 ----
#pragma unroll
    for (int nt = 0; nt < 4; ++nt)
#pragma unroll
      for (int r = 0; r < 8; ++r) {
        float th = fast_tanh_biased(acc[nt][r], tb1[nt]);
        swh[(dstBase + r) * 32 + 8 * nt + sub] = (_Float16)th;
      }
    v16h a2_0 = *(const v16h*)(swh + lane * 32);        // contiguous 32B (in-order DS)
    v16h a2_1 = *(const v16h*)(swh + lane * 32 + 16);

    // ---- GEMM2: h = tanh(...)@W2, B tiles streamed from LDS, C=0 ----
    v8f h4[4];
#pragma unroll
    for (int nt = 0; nt < 4; ++nt) {
      v16h bb0 = *(const v16h*)&w2lds[((0 * 4 + nt) * 32 + lane) * 16];
      v16h bb1 = *(const v16h*)&w2lds[((1 * 4 + nt) * 32 + lane) * 16];
      v8f a = wmma_f16(a2_0, bb0, zacc);
      a     = wmma_f16(a2_1, bb1, a);
      h4[nt] = a;
    }

    // ---- D->A transpose of h (f32), folding in b2 on the way ----
#pragma unroll
    for (int nt = 0; nt < 4; ++nt)
#pragma unroll
      for (int r = 0; r < 8; ++r)
        sw[(dstBase + r) * 32 + 8 * nt + sub] = h4[nt][r] + b2v[nt];
    float h[32];
#pragma unroll
    for (int j = 0; j < 8; ++j) {
      float4 v = *(const float4*)(sw + lane * 32 + 4 * j);
      h[4 * j + 0] = v.x; h[4 * j + 1] = v.y; h[4 * j + 2] = v.z; h[4 * j + 3] = v.w;
    }

    // ---- complex-pair rotation: pairs (q, q+1) are lane-local by construction ----
#pragma unroll
    for (int q = 0; q < 32; q += 2) {
      float ev = __expf(DT * h[q]);       // v_exp_f32
      float cc = __cosf(DT * h[q + 1]);   // v_cos_f32
      float ss = __sinf(DT * h[q + 1]);   // v_sin_f32
      float yr = y[q], yi = y[q + 1];
      y[q]     = ev * (cc * yr - ss * yi);
      y[q + 1] = ev * (ss * yr + cc * yi);
    }

    // ---- mandatory output stores: 8 contiguous floats per (lane, n-tile) ----
    float* op = out + ((size_t)myRow * T_STEPS + step) * K_SZ;
#pragma unroll
    for (int t = 0; t < 4; ++t) {
      float4 v0 = { y[8 * t + 0], y[8 * t + 1], y[8 * t + 2], y[8 * t + 3] };
      float4 v1 = { y[8 * t + 4], y[8 * t + 5], y[8 * t + 6], y[8 * t + 7] };
      *(float4*)(op + 16 * t + 8 * g)     = v0;   // global_store_b128
      *(float4*)(op + 16 * t + 8 * g + 4) = v1;
    }
  }
}

extern "C" void kernel_launch(void* const* d_in, const int* in_sizes, int n_in,
                              void* d_out, int out_size, void* d_ws, size_t ws_size,
                              hipStream_t stream) {
  const float* x  = (const float*)d_in[0];
  const float* W1 = (const float*)d_in[1];
  const float* b1 = (const float*)d_in[2];
  const float* W2 = (const float*)d_in[3];
  const float* b2 = (const float*)d_in[4];
  float* out = (float*)d_out;

  dim3 grid(B_SZ / (16 * WAVES));   // 64 blocks
  dim3 block(32 * WAVES);           // 128 threads = 4 waves, 16 rows each
  hipLaunchKernelGGL(koopman_kernel, grid, block, 0, stream, x, W1, b1, W2, b2, out);
}
